// MultiHeadAttention_9887014716111
// MI455X (gfx1250) — matrix operand order repaired and checked by reading, not compiled, not run on hardware
//
#include <hip/hip_runtime.h>
#include <hip/hip_bf16.h>
#include <cstdint>

// MHA forward for gfx1250 (MI455X): bf16 WMMA GEMMs with double-buffered
// TDM (tensor_load_to_lds) staging of A and W tiles + flash attention.
// B=4, S=2048, E=1024, H=16, D=64.

#define B_ 4
#define S_ 2048
#define E_ 1024
#define H_ 16
#define D_ 64
#define NROWS_ (B_ * S_)   // 8192

typedef __bf16 bf16_t;
typedef __attribute__((ext_vector_type(16))) __bf16 v16bf;
typedef __attribute__((ext_vector_type(8)))  __bf16 v8bf;
typedef __attribute__((ext_vector_type(8)))  float  v8f;
typedef __attribute__((ext_vector_type(4)))  unsigned int u32x4;
typedef __attribute__((ext_vector_type(4)))  int i32x4;
typedef __attribute__((ext_vector_type(8)))  int i32x8;

#if defined(__has_builtin)
#if __has_builtin(__builtin_amdgcn_tensor_load_to_lds)
#define HAVE_TDM 1
#endif
#endif

// Build a 16-element bf16 fragment from two contiguous 16-byte chunks.
// A order used here (matched to B): lane m = lane&15; elems e=0..15 -> K = half*16 + e
// (half = lane>>4).
// B order used here: lane n = lane&15; elems e=0..15 -> K = half*16 + e.
// Both reduce to two aligned b128 loads from row-major memory (global or LDS).
template <typename P>
__device__ __forceinline__ v16bf ld_frag2(P lo, P hi) {
    v8bf a = *(const v8bf*)lo;
    v8bf b = *(const v8bf*)hi;
    v16bf r;
#pragma unroll
    for (int i = 0; i < 8; ++i) { r[i] = a[i]; r[i + 8] = b[i]; }
    return r;
}

__device__ __forceinline__ v8f wmma_bf16(v16bf a, v16bf b, v8f c) {
    return __builtin_amdgcn_wmma_f32_16x16x32_bf16(
        /*neg_a=*/false, a, /*neg_b=*/false, b,
        /*c_mod=*/(short)0, c, /*reuse_a=*/false, /*reuse_b=*/false);
}

__device__ __forceinline__ float rmax16(float x) {
    x = fmaxf(x, __shfl_xor(x, 1, 32));
    x = fmaxf(x, __shfl_xor(x, 2, 32));
    x = fmaxf(x, __shfl_xor(x, 4, 32));
    x = fmaxf(x, __shfl_xor(x, 8, 32));
    return x;
}
__device__ __forceinline__ float rsum16(float x) {
    x += __shfl_xor(x, 1, 32);
    x += __shfl_xor(x, 2, 32);
    x += __shfl_xor(x, 4, 32);
    x += __shfl_xor(x, 8, 32);
    return x;
}

#ifdef HAVE_TDM
// Issue one TDM 2D tile load: tile (tile0 x tile1) of 2-byte elements from a
// row-major tensor (dims td0 x td1, line stride stride0) into LDS at lds_addr.
// D# layout per CDNA5 ISA ch.8 (group0: control/lds/global addr; group1: dims).
__device__ __forceinline__ void tdm_load_2d_b16(unsigned lds_addr, const void* gptr,
                                                unsigned td0, unsigned td1,
                                                unsigned tile0, unsigned tile1,
                                                unsigned stride0) {
    uint64_t ga = (uint64_t)gptr;
    u32x4 g0;
    g0[0] = 1u;                                              // count=1, user D#
    g0[1] = lds_addr;                                        // lds_addr[31:0]
    g0[2] = (unsigned)ga;                                    // global_addr[31:0]
    g0[3] = (unsigned)((ga >> 32) & 0x01FFFFFFu) | (2u << 30); // addr[56:32] | type=2
    i32x8 g1;
    g1[0] = (int)(1u << 16);                                 // data_size=1 (2B), mask=0
    g1[1] = (int)((td0 & 0xFFFFu) << 16);                    // tensor_dim0[15:0]
    g1[2] = (int)((td0 >> 16) | ((td1 & 0xFFFFu) << 16));    // dim0[31:16] | dim1[15:0]
    g1[3] = (int)((td1 >> 16) | (tile0 << 16));              // dim1[31:16] | tile_dim0
    g1[4] = (int)(tile1 & 0xFFFFu);                          // tile_dim1, tile_dim2=0
    g1[5] = (int)stride0;                                    // dim0_stride[31:0]
    g1[6] = 0;                                               // dim0_stride[47:32], dim1_stride lo
    g1[7] = 0;
    i32x4 z4 = {};
#if __clang_major__ >= 23
    i32x8 z8 = {};
    __builtin_amdgcn_tensor_load_to_lds(g0, g1, z4, z4, z8, 0);
#else
    __builtin_amdgcn_tensor_load_to_lds(g0, g1, z4, z4, 0);
#endif
}
#endif

// ---------------------------------------------------------------- converts
__global__ void mha_cvt_f32_bf16(const float* __restrict__ in,
                                 bf16_t* __restrict__ out, int n) {
    int i = blockIdx.x * blockDim.x + threadIdx.x;
    if (i < n) out[i] = (bf16_t)in[i];
}

// ---------------------------------------------------------------- GEMM
// out = A @ W^T + bias.  A: [M,K] bf16 row-major. W: [Ncols,K] bf16 row-major.
// Block (256 thr, 8 waves) computes a 128x128 output tile.  Per K-step the
// 128x32 A tile and 128x32 W tile are DMA'd into LDS by the Tensor Data Mover,
// double-buffered so the DMA for step i+1 overlaps the WMMAs of step i.
// Each wave computes 16 rows x 128 cols (one A-fragment feeds 8 WMMAs).
// mode 0: store bf16 into [B,H,S,D]   (q, k)
// mode 1: store bf16 into [B,H,D,S]   (v, pre-transposed for PV gemm)
// mode 2: store f32 into [M, E]       (final output, d_out)
// grid: (M/128, Ncols/128)
__global__ __launch_bounds__(256)
void mha_gemm_wmma(const bf16_t* __restrict__ A,
                   const bf16_t* __restrict__ W,
                   const float* __restrict__ bias,
                   void* __restrict__ out,
                   int M, int K, int mode) {
    __shared__ bf16_t atile[2][128 * 32];   // A tile  [m:128][k:32], 8 KB x2
    __shared__ bf16_t btile[2][128 * 32];   // W tile  [o:128][k:32], 8 KB x2

    const int lane = threadIdx.x & 31;
    const int wave = threadIdx.x >> 5;
    const int hl   = lane >> 4;
    const int l16  = lane & 15;

    const int mb = blockIdx.x * 128;
    const int ob = blockIdx.y * 128;
    const int mw = mb + wave * 16;          // this wave's 16 output rows

    v8f acc[8];
#pragma unroll
    for (int t = 0; t < 8; ++t) acc[t] = (v8f){};

#ifdef HAVE_TDM
    // prologue: stage first A/W tiles into buffer 0
    if (wave == 0) {
        tdm_load_2d_b16((unsigned)(uintptr_t)&atile[0][0],
                        (const void*)(A + (size_t)mb * K),
                        (unsigned)K, (unsigned)M, 32u, 128u, (unsigned)K);
        tdm_load_2d_b16((unsigned)(uintptr_t)&btile[0][0],
                        (const void*)(W + (size_t)ob * K),
                        (unsigned)K, (unsigned)E_, 32u, 128u, (unsigned)K);
    }
#endif

    for (int k0 = 0, it = 0; k0 < K; k0 += 32, ++it) {
        const int cur = it & 1;

#ifdef HAVE_TDM
        if (wave == 0) {
            __builtin_amdgcn_s_wait_tensorcnt((short)0);   // buffer `cur` ready
        }
        __syncthreads();   // all waves: tiles staged, prior reads of `cur^1` done
        if (wave == 0 && k0 + 32 < K) {                    // prefetch next K-step
            tdm_load_2d_b16((unsigned)(uintptr_t)&atile[cur ^ 1][0],
                            (const void*)(A + (size_t)mb * K + k0 + 32),
                            (unsigned)K, (unsigned)M, 32u, 128u, (unsigned)K);
            tdm_load_2d_b16((unsigned)(uintptr_t)&btile[cur ^ 1][0],
                            (const void*)(W + (size_t)ob * K + k0 + 32),
                            (unsigned)K, (unsigned)E_, 32u, 128u, (unsigned)K);
        }
#else
        {   // cooperative copy fallback: 256 thr x 16 elems per tile
            int t = threadIdx.x;
#pragma unroll
            for (int i = 0; i < 16; ++i) {
                int idx = t * 16 + i;
                atile[cur][idx] = A[(size_t)(mb + (idx >> 5)) * K + k0 + (idx & 31)];
                btile[cur][idx] = W[(size_t)(ob + (idx >> 5)) * K + k0 + (idx & 31)];
            }
        }
        __syncthreads();
#endif

        const bf16_t* arow = &atile[cur][(wave * 16 + l16) * 32];
        v16bf a = ld_frag2(arow + hl * 16, arow + hl * 16 + 8);
#pragma unroll
        for (int t = 0; t < 8; ++t) {
            const bf16_t* brow = &btile[cur][(t * 16 + l16) * 32 + hl * 16];
            v16bf b = ld_frag2(brow, brow + 8);
            acc[t] = wmma_bf16(a, b, acc[t]);
        }
#ifndef HAVE_TDM
        __syncthreads();
#endif
    }

    if (mode == 2) {
        float* of = (float*)out;
#pragma unroll
        for (int t = 0; t < 8; ++t) {
            float bcol = bias[ob + t * 16 + l16];
#pragma unroll
            for (int r = 0; r < 8; ++r) {
                int gr = mw + hl * 8 + r;       // C layout: row = hl*8+r, col = l16
                of[(size_t)gr * E_ + ob + t * 16 + l16] = acc[t][r] + bcol;
            }
        }
    } else {
        bf16_t* ob_ptr = (bf16_t*)out;
#pragma unroll
        for (int t = 0; t < 8; ++t) {
            const int o = ob + t * 16 + l16;
            const int h = o >> 6, d = o & 63;
            float bcol = bias[o];
#pragma unroll
            for (int r = 0; r < 8; ++r) {
                int gr = mw + hl * 8 + r;
                int b  = gr >> 11;              // S_ = 2048
                int s  = gr & (S_ - 1);
                size_t idx = (mode == 0)
                    ? ((size_t)(b * H_ + h) * S_ + s) * D_ + d
                    : ((size_t)(b * H_ + h) * D_ + d) * S_ + s;
                ob_ptr[idx] = (bf16_t)(acc[t][r] + bcol);
            }
        }
    }
}

// ---------------------------------------------------------------- attention
// q,k: [B*H, S, D] bf16.  vt: [B*H, D, S] bf16.  attn_out: [NROWS, E] bf16.
// One wave per 16-query tile; flash (online softmax) over key blocks of 32.
__global__ __launch_bounds__(256)
void mha_attn_wmma(const bf16_t* __restrict__ q,
                   const bf16_t* __restrict__ k,
                   const bf16_t* __restrict__ vt,
                   bf16_t* __restrict__ attn_out) {
    __shared__ bf16_t lds_p[8][16 * 32];   // per-wave P-tile staging (C->A relayout)

    const int lane = threadIdx.x & 31;
    const int wave = threadIdx.x >> 5;
    const int hl   = lane >> 4;
    const int l16  = lane & 15;

    const int gtile = blockIdx.x * 8 + wave;   // 0..8191
    const int qt = gtile & 127;                // q tile within (b,h)
    const int bh = gtile >> 7;                 // 0..63
    const int q0 = qt * 16;

    const bf16_t* qbase = q  + ((size_t)bh * S_ + q0) * D_;
    const bf16_t* kbase = k  + (size_t)bh * S_ * D_;
    const bf16_t* vbase = vt + (size_t)bh * D_ * S_;

    // Q A-fragments (rows m = l16, K = d), two K-steps covering D=64.
    const bf16_t* qrow = qbase + (size_t)l16 * D_;
    v16bf aq0 = ld_frag2(qrow + hl * 16,      qrow + hl * 16 + 8);
    v16bf aq1 = ld_frag2(qrow + 32 + hl * 16, qrow + 32 + hl * 16 + 8);

    v8f acc0 = {}, acc1 = {}, acc2 = {}, acc3 = {};
    float m_i[8], l_i[8];
#pragma unroll
    for (int r = 0; r < 8; ++r) { m_i[r] = -__builtin_inff(); l_i[r] = 0.0f; }

    const float SCALE = 0.125f;   // 1/sqrt(64)

    for (int kb0 = 0; kb0 < S_; kb0 += 32) {
        // ---- scores: two 16x16 tiles (keys kb0..+15, kb0+16..+31)
        v8f s0 = {}, s1 = {};
        {
            const bf16_t* k0row = kbase + (size_t)(kb0 + l16) * D_;
            const bf16_t* k1row = kbase + (size_t)(kb0 + 16 + l16) * D_;
            v16bf bk;
            bk = ld_frag2(k0row + hl * 16,      k0row + hl * 16 + 8);
            s0 = wmma_bf16(aq0, bk, s0);
            bk = ld_frag2(k0row + 32 + hl * 16, k0row + 32 + hl * 16 + 8);
            s0 = wmma_bf16(aq1, bk, s0);
            bk = ld_frag2(k1row + hl * 16,      k1row + hl * 16 + 8);
            s1 = wmma_bf16(aq0, bk, s1);
            bk = ld_frag2(k1row + 32 + hl * 16, k1row + 32 + hl * 16 + 8);
            s1 = wmma_bf16(aq1, bk, s1);
        }

        // ---- online softmax update (per output row m = hl*8 + r)
#pragma unroll
        for (int r = 0; r < 8; ++r) {
            float v0 = s0[r] * SCALE;
            float v1 = s1[r] * SCALE;
            float mx   = rmax16(fmaxf(v0, v1));
            float mnew = fmaxf(m_i[r], mx);
            float alpha = __expf(m_i[r] - mnew);
            float p0 = __expf(v0 - mnew);
            float p1 = __expf(v1 - mnew);
            float ps = rsum16(p0 + p1);
            l_i[r] = l_i[r] * alpha + ps;
            m_i[r] = mnew;
            acc0[r] *= alpha; acc1[r] *= alpha; acc2[r] *= alpha; acc3[r] *= alpha;
            // stage P (C layout -> LDS row-major 16x32)
            lds_p[wave][(hl * 8 + r) * 32 + l16]      = (bf16_t)p0;
            lds_p[wave][(hl * 8 + r) * 32 + 16 + l16] = (bf16_t)p1;
        }
        asm volatile("s_wait_dscnt 0x0" ::: "memory");

        // re-read P as an A fragment (rows m = l16, K = key-in-block)
        const bf16_t* prow = &lds_p[wave][l16 * 32];
        v16bf ap = ld_frag2(prow + hl * 16, prow + hl * 16 + 8);

        // ---- PV: vt is [D, S], so B fragments are contiguous along s.
        {
            const bf16_t* vr;
            v16bf bv;
            vr = vbase + (size_t)(0 * 16 + l16) * S_ + kb0 + hl * 16;
            bv = ld_frag2(vr, vr + 8);  acc0 = wmma_bf16(ap, bv, acc0);
            vr = vbase + (size_t)(1 * 16 + l16) * S_ + kb0 + hl * 16;
            bv = ld_frag2(vr, vr + 8);  acc1 = wmma_bf16(ap, bv, acc1);
            vr = vbase + (size_t)(2 * 16 + l16) * S_ + kb0 + hl * 16;
            bv = ld_frag2(vr, vr + 8);  acc2 = wmma_bf16(ap, bv, acc2);
            vr = vbase + (size_t)(3 * 16 + l16) * S_ + kb0 + hl * 16;
            bv = ld_frag2(vr, vr + 8);  acc3 = wmma_bf16(ap, bv, acc3);
        }
    }

    // ---- finalize: attn_out[b*S + row, h*64 + d]
    const int b = bh >> 4;
    const int h = bh & 15;
#pragma unroll
    for (int r = 0; r < 8; ++r) {
        float inv = 1.0f / l_i[r];
        int row = q0 + hl * 8 + r;
        size_t base = ((size_t)b * S_ + row) * E_ + h * 64 + l16;
        attn_out[base + 0]  = (bf16_t)(acc0[r] * inv);
        attn_out[base + 16] = (bf16_t)(acc1[r] * inv);
        attn_out[base + 32] = (bf16_t)(acc2[r] * inv);
        attn_out[base + 48] = (bf16_t)(acc3[r] * inv);
    }
}

// ---------------------------------------------------------------- launch
extern "C" void kernel_launch(void* const* d_in, const int* in_sizes, int n_in,
                              void* d_out, int out_size, void* d_ws, size_t ws_size,
                              hipStream_t stream) {
    (void)in_sizes; (void)n_in; (void)out_size; (void)ws_size;

    const float* x  = (const float*)d_in[0];
    const float* wq = (const float*)d_in[1];
    const float* bq = (const float*)d_in[2];
    const float* wk = (const float*)d_in[3];
    const float* bk = (const float*)d_in[4];
    const float* wv = (const float*)d_in[5];
    const float* bv = (const float*)d_in[6];
    const float* wo = (const float*)d_in[7];
    const float* bo = (const float*)d_in[8];

    // workspace layout (bf16): x, 4 weights, q, k, vt, attn  => 88 MB total
    char* ws = (char*)d_ws;
    size_t off = 0;
    bf16_t* x_bf  = (bf16_t*)(ws + off); off += (size_t)NROWS_ * E_ * sizeof(bf16_t);
    bf16_t* wq_bf = (bf16_t*)(ws + off); off += (size_t)E_ * E_ * sizeof(bf16_t);
    bf16_t* wk_bf = (bf16_t*)(ws + off); off += (size_t)E_ * E_ * sizeof(bf16_t);
    bf16_t* wv_bf = (bf16_t*)(ws + off); off += (size_t)E_ * E_ * sizeof(bf16_t);
    bf16_t* wo_bf = (bf16_t*)(ws + off); off += (size_t)E_ * E_ * sizeof(bf16_t);
    bf16_t* q_bf  = (bf16_t*)(ws + off); off += (size_t)NROWS_ * E_ * sizeof(bf16_t);
    bf16_t* k_bf  = (bf16_t*)(ws + off); off += (size_t)NROWS_ * E_ * sizeof(bf16_t);
    bf16_t* vt_bf = (bf16_t*)(ws + off); off += (size_t)NROWS_ * E_ * sizeof(bf16_t);
    bf16_t* at_bf = (bf16_t*)(ws + off); off += (size_t)NROWS_ * E_ * sizeof(bf16_t);

    const int nX = NROWS_ * E_;
    const int nW = E_ * E_;
    mha_cvt_f32_bf16<<<(nX + 255) / 256, 256, 0, stream>>>(x,  x_bf,  nX);
    mha_cvt_f32_bf16<<<(nW + 255) / 256, 256, 0, stream>>>(wq, wq_bf, nW);
    mha_cvt_f32_bf16<<<(nW + 255) / 256, 256, 0, stream>>>(wk, wk_bf, nW);
    mha_cvt_f32_bf16<<<(nW + 255) / 256, 256, 0, stream>>>(wv, wv_bf, nW);
    mha_cvt_f32_bf16<<<(nW + 255) / 256, 256, 0, stream>>>(wo, wo_bf, nW);

    dim3 gg(NROWS_ / 128, E_ / 128);
    mha_gemm_wmma<<<gg, 256, 0, stream>>>(x_bf, wq_bf, bq, q_bf,  NROWS_, E_, 0);
    mha_gemm_wmma<<<gg, 256, 0, stream>>>(x_bf, wk_bf, bk, k_bf,  NROWS_, E_, 0);
    mha_gemm_wmma<<<gg, 256, 0, stream>>>(x_bf, wv_bf, bv, vt_bf, NROWS_, E_, 1);

    const int nTiles = B_ * H_ * (S_ / 16);          // 8192 wave-tiles
    mha_attn_wmma<<<nTiles / 8, 256, 0, stream>>>(q_bf, k_bf, vt_bf, at_bf);

    mha_gemm_wmma<<<gg, 256, 0, stream>>>(at_bf, wo_bf, bo, d_out, NROWS_, E_, 2);
}
